// MoELayer_14860586844370
// MI455X (gfx1250) — compile-verified, matrix operand order checked by reading, not run on hardware
//
#include <hip/hip_runtime.h>
#include <hip/hip_bf16.h>
#include <math.h>

// ---------------------------------------------------------------------------
// MoE (shared + 8 routed experts, top-2) for MI455X / gfx1250.
// Compute-bound -> all GEMM math via v_wmma_f32_16x16x32_bf16 with double-
// buffered LDS tiles (one barrier per K-step).  Routed experts run on
// COMPACTED token lists (top-2 of 8 => 3x fewer FLOPs than the dense loop:
// ~141 GF instead of ~425 GF), which is the dominant lever on this chip.
// Unselected experts have exactly-zero combine weight in the reference, so
// skipping them is numerically identical.  Atomic list build only permutes
// slots; per-token results are slot-invariant and experts accumulate in fixed
// launch order -> deterministic output.
// ---------------------------------------------------------------------------

typedef __attribute__((ext_vector_type(16))) __bf16 v16bf;
typedef __attribute__((ext_vector_type(8)))  float  v8f;

constexpr int Tn  = 4 * 2048;   // tokens
constexpr int Dn  = 1024;       // hidden
constexpr int En  = 8;          // experts
constexpr int In  = 938;        // expert_inter
constexpr int IPn = 960;        // In padded to multiple of 32 (WMMA K)

#define LDSLD 40                // LDS row stride (bf16): 32 data + 8 pad

// ---- WMMA fragment load from an LDS tile (bf16, row-major, stride LDSLD) ---
// 16-bit A/B layout (ISA 7.12.2): lane L<16 -> row L, K {kb..kb+7, 16+kb..};
// lanes 16..31 -> row L-16 with kb=8.  Two aligned 16-byte LDS reads.
__device__ __forceinline__ v16bf load_frag(const __bf16* base) {
  const int lane = threadIdx.x & 31;
  const __bf16* p = base + (lane & 15) * LDSLD + ((lane >> 4) << 3);
  union { v16bf v; uint4 q[2]; } u;
  u.q[0] = *(const uint4*)(p);
  u.q[1] = *(const uint4*)(p + 16);
  return u.v;
}

// ---------------------------------------------------------------------------
__global__ void init_counts_kernel(int* __restrict__ cnt) {
  if (threadIdx.x < En) cnt[threadIdx.x] = 0;
}

// ---------------------------------------------------------------------------
// Router: one wave per token.  logits -> softmax -> top2 -> normalized
// weights, appended to per-expert compacted token lists.
// ---------------------------------------------------------------------------
__global__ __launch_bounds__(256)
void router_kernel(const float* __restrict__ x, const float* __restrict__ gw,
                   int* __restrict__ cnt, int* __restrict__ idx,
                   float* __restrict__ wts) {
  const int lane = threadIdx.x & 31;
  const int t    = blockIdx.x * 8 + (threadIdx.x >> 5);
  const float* xr = x + (size_t)t * Dn;

  float acc[En];
#pragma unroll
  for (int e = 0; e < En; ++e) acc[e] = 0.f;

  for (int d = lane; d < Dn; d += 32) {
    const float xv = xr[d];
#pragma unroll
    for (int e = 0; e < En; ++e) acc[e] += xv * gw[e * Dn + d];
  }
#pragma unroll
  for (int e = 0; e < En; ++e)
#pragma unroll
    for (int off = 16; off > 0; off >>= 1) acc[e] += __shfl_xor(acc[e], off, 32);

  if (lane == 0) {
    float mx = acc[0];
#pragma unroll
    for (int e = 1; e < En; ++e) mx = fmaxf(mx, acc[e]);
    float p[En], s = 0.f;
#pragma unroll
    for (int e = 0; e < En; ++e) { p[e] = __expf(acc[e] - mx); s += p[e]; }
    const float inv = 1.f / s;
#pragma unroll
    for (int e = 0; e < En; ++e) p[e] *= inv;
    int i1 = 0;                              // first index wins ties (lax.top_k)
#pragma unroll
    for (int e = 1; e < En; ++e) if (p[e] > p[i1]) i1 = e;
    int i2 = (i1 == 0) ? 1 : 0;
#pragma unroll
    for (int e = 0; e < En; ++e) if (e != i1 && p[e] > p[i2]) i2 = e;
    const float d2 = p[i1] + p[i2] + 1e-8f;

    const int p1 = atomicAdd(&cnt[i1], 1);
    idx[i1 * Tn + p1] = t;  wts[i1 * Tn + p1] = p[i1] / d2;
    const int p2 = atomicAdd(&cnt[i2], 1);
    idx[i2 * Tn + p2] = t;  wts[i2 * Tn + p2] = p[i2] / d2;
  }
}

// ---------------------------------------------------------------------------
// fp32 -> bf16 cast of x (done once; reused by all GEMM passes)
// ---------------------------------------------------------------------------
__global__ __launch_bounds__(256)
void cast_kernel(const float* __restrict__ x, __bf16* __restrict__ xb) {
  const size_t i = ((size_t)blockIdx.x * 256 + threadIdx.x) * 8;
  float4 a = *(const float4*)(x + i);
  float4 b = *(const float4*)(x + i + 4);
  union { __bf16 h[8]; uint4 v; } t;
  t.h[0] = (__bf16)a.x; t.h[1] = (__bf16)a.y; t.h[2] = (__bf16)a.z; t.h[3] = (__bf16)a.w;
  t.h[4] = (__bf16)b.x; t.h[5] = (__bf16)b.y; t.h[6] = (__bf16)b.z; t.h[7] = (__bf16)b.w;
  *(uint4*)(xb + i) = t.v;
}

// ---------------------------------------------------------------------------
// Pass 1: act[slot,i] = silu(x[tok]@Wg[i]) * (x[tok]@Wu[i])  (bf16, [.,IPn])
// tok = idxp[slot] (compacted list) or slot (shared pass, idxp==null).
// Block tile 128(M) x 64(N), K-step 32, double-buffered LDS, 1 barrier/step.
// Blocks beyond the compacted count early-exit; partial blocks pad with a
// valid token so every act row they own is written with defined values.
// ---------------------------------------------------------------------------
__global__ __launch_bounds__(256)
void gemm_act_kernel(const __bf16* __restrict__ xb, const float* __restrict__ wgu,
                     __bf16* __restrict__ act, const int* __restrict__ idxp,
                     const int* __restrict__ cntp) {
  const int m0 = blockIdx.y * 128;
  const int cnt = cntp ? *cntp : Tn;
  if (m0 >= cnt) return;

  __shared__ __bf16 As[2][128 * LDSLD];
  __shared__ __bf16 Bg[2][64 * LDSLD];
  __shared__ __bf16 Bu[2][64 * LDSLD];

  const int tid  = threadIdx.x;
  const int lane = tid & 31;
  const int wv   = tid >> 5, wm = wv >> 2, wn = wv & 3;
  const int n0   = blockIdx.x * 64;          // padded inter space [0,960)

  // loader coordinates (fixed across K-steps)
  const int ar = tid >> 1, ah = (tid & 1) * 16;   // A: 128 rows x 2 half-rows
  const int br = tid >> 2, bq = (tid & 3) * 8;    // B: 64 rows x 4 octets
  const int slot = m0 + ar;
  const int tok  = idxp ? idxp[(slot < cnt) ? slot : m0] : slot;
  const int bn = n0 + br;
  const bool bOk = (bn < In);
  const __bf16* aSrc = xb + (size_t)tok * Dn + ah;
  const float*  gSrc = wgu + (size_t)bn * Dn + bq;          // valid iff bOk
  const float*  uSrc = wgu + (size_t)(In + bn) * Dn + bq;
  const int aDst = ar * LDSLD + ah;
  const int bDst = br * LDSLD + bq;

  // guarded pad rows (n>=938): zero once in both buffers, never rewritten
  if (!bOk) {
    const uint4 z = {0u, 0u, 0u, 0u};
    *(uint4*)&Bg[0][bDst] = z; *(uint4*)&Bg[1][bDst] = z;
    *(uint4*)&Bu[0][bDst] = z; *(uint4*)&Bu[1][bDst] = z;
  }

  uint4 a0, a1;
  union { __bf16 h[8]; uint4 v; } tg, tu;

  // prologue: stage K-tile 0
  a0 = *(const uint4*)(aSrc);
  a1 = *(const uint4*)(aSrc + 8);
  if (bOk) {
#pragma unroll
    for (int j = 0; j < 8; ++j) { tg.h[j] = (__bf16)gSrc[j]; tu.h[j] = (__bf16)uSrc[j]; }
  }
  *(uint4*)&As[0][aDst]     = a0;
  *(uint4*)&As[0][aDst + 8] = a1;
  if (bOk) { *(uint4*)&Bg[0][bDst] = tg.v; *(uint4*)&Bu[0][bDst] = tu.v; }
  __syncthreads();

  v8f accg[4] = {}, accu[4] = {};
  constexpr int KT = Dn / 32;
  for (int kt = 0; kt < KT; ++kt) {
    const int  c    = kt & 1;
    const bool more = (kt + 1 < KT);
    if (more) {                               // stage next tile into registers
      const int k1 = (kt + 1) * 32;
      a0 = *(const uint4*)(aSrc + k1);
      a1 = *(const uint4*)(aSrc + k1 + 8);
      if (bOk) {
        __builtin_prefetch(gSrc + k1 + 32, 0, 1);
        __builtin_prefetch(uSrc + k1 + 32, 0, 1);
#pragma unroll
        for (int j = 0; j < 8; ++j) {
          tg.h[j] = (__bf16)gSrc[k1 + j];
          tu.h[j] = (__bf16)uSrc[k1 + j];
        }
      }
    }
    // math on current buffer
    const v16bf bg = load_frag(&Bg[c][(wn * 16) * LDSLD]);
    const v16bf bu = load_frag(&Bu[c][(wn * 16) * LDSLD]);
    const __bf16* Ab = &As[c][(wm * 64) * LDSLD];
#pragma unroll
    for (int s = 0; s < 4; ++s) {
      const v16bf a = load_frag(Ab + (s * 16) * LDSLD);
      accg[s] = __builtin_amdgcn_wmma_f32_16x16x32_bf16(false, a, false, bg,
                                                        (short)0, accg[s], false, false);
      accu[s] = __builtin_amdgcn_wmma_f32_16x16x32_bf16(false, a, false, bu,
                                                        (short)0, accu[s], false, false);
    }
    if (more) {                               // commit staged tile, 1 barrier
      const int nb = c ^ 1;
      *(uint4*)&As[nb][aDst]     = a0;
      *(uint4*)&As[nb][aDst + 8] = a1;
      if (bOk) { *(uint4*)&Bg[nb][bDst] = tg.v; *(uint4*)&Bu[nb][bDst] = tu.v; }
      __syncthreads();
    }
  }

  // epilogue: silu(g)*u, bf16 store (pad cols exact 0; pad slots defined)
  const int cn = lane & 15;
  const int mh = (lane >> 4) << 3;
  const int nn = n0 + wn * 16 + cn;
#pragma unroll
  for (int s = 0; s < 4; ++s) {
    const int mb = m0 + wm * 64 + s * 16 + mh;
#pragma unroll
    for (int r = 0; r < 8; ++r) {
      const float g = accg[s][r];
      const float v = (g * (1.f / (1.f + __expf(-g)))) * accu[s][r];
      act[(size_t)(mb + r) * IPn + nn] = (__bf16)v;
    }
  }
}

// ---------------------------------------------------------------------------
// Pass 2: y = act @ down^T ; scatter out[tok] (+)= w * y
// M = compacted count (slot-indexed act rows, no gather on A), N=D, K=IPn.
// Block tile 128x128, double-buffered LDS; 8 waves = 2(M) x 4(N), each wave
// 4 M-subtiles x 2 N-subtiles (8 WMMAs per K-step / 6 fragment loads).
// ---------------------------------------------------------------------------
__global__ __launch_bounds__(256)
void gemm_down_kernel(const __bf16* __restrict__ act, const float* __restrict__ wd,
                      float* __restrict__ out, const int* __restrict__ idxp,
                      const float* __restrict__ wtsp, const int* __restrict__ cntp) {
  const int m0 = blockIdx.y * 128;
  const int cnt = cntp ? *cntp : Tn;
  if (m0 >= cnt) return;

  __shared__ __bf16 As[2][128 * LDSLD];
  __shared__ __bf16 Bs[2][128 * LDSLD];

  const int tid  = threadIdx.x;
  const int lane = tid & 31;
  const int wv   = tid >> 5, wm = wv >> 2, wn = wv & 3;
  const int n0   = blockIdx.x * 128;          // d base

  // loader coordinates: both tiles are 128 rows x 32 cols, 16 elems/thread
  const int lr = tid >> 1, lh = (tid & 1) * 16;
  const __bf16* aSrc = act + (size_t)(m0 + lr) * IPn + lh;
  const float*  bSrc = wd + (size_t)(n0 + lr) * In + lh;   // row always valid
  const int dst = lr * LDSLD + lh;

  uint4 a0, a1;
  union { __bf16 h[16]; uint4 v[2]; } tb;

  auto loadB = [&](int k0) {
    const int kg = k0 + lh;
    if (kg + 15 < In) {                       // fast path (all but last K-tile)
#pragma unroll
      for (int j = 0; j < 16; ++j) tb.h[j] = (__bf16)bSrc[k0 + j];
    } else {
#pragma unroll
      for (int j = 0; j < 16; ++j)
        tb.h[j] = (__bf16)((kg + j < In) ? bSrc[k0 + j] : 0.f);
    }
  };

  // prologue: stage K-tile 0
  a0 = *(const uint4*)(aSrc);
  a1 = *(const uint4*)(aSrc + 8);
  loadB(0);
  *(uint4*)&As[0][dst]     = a0;
  *(uint4*)&As[0][dst + 8] = a1;
  *(uint4*)&Bs[0][dst]     = tb.v[0];
  *(uint4*)&Bs[0][dst + 8] = tb.v[1];
  __syncthreads();

  v8f acc[4][2] = {};
  constexpr int KT = IPn / 32;
  for (int kt = 0; kt < KT; ++kt) {
    const int  c    = kt & 1;
    const bool more = (kt + 1 < KT);
    if (more) {
      const int k1 = (kt + 1) * 32;
      a0 = *(const uint4*)(aSrc + k1);
      a1 = *(const uint4*)(aSrc + k1 + 8);
      __builtin_prefetch(bSrc + k1 + 32, 0, 1);
      loadB(k1);
    }
    v16bf bf[2];
#pragma unroll
    for (int t = 0; t < 2; ++t)
      bf[t] = load_frag(&Bs[c][(wn * 32 + t * 16) * LDSLD]);
    const __bf16* Ab = &As[c][(wm * 64) * LDSLD];
#pragma unroll
    for (int s = 0; s < 4; ++s) {
      const v16bf a = load_frag(Ab + (s * 16) * LDSLD);
#pragma unroll
      for (int t = 0; t < 2; ++t)
        acc[s][t] = __builtin_amdgcn_wmma_f32_16x16x32_bf16(false, a, false, bf[t],
                                                            (short)0, acc[s][t], false, false);
    }
    if (more) {
      const int nb = c ^ 1;
      *(uint4*)&As[nb][dst]     = a0;
      *(uint4*)&As[nb][dst + 8] = a1;
      *(uint4*)&Bs[nb][dst]     = tb.v[0];
      *(uint4*)&Bs[nb][dst + 8] = tb.v[1];
      __syncthreads();
    }
  }

  // epilogue: shared pass (idxp==null): out[slot] = y
  //           routed pass: slot<cnt -> out[idx[slot]] += wts[slot] * y
  const int cn = lane & 15;
  const int mh = (lane >> 4) << 3;
#pragma unroll
  for (int t = 0; t < 2; ++t) {
    const int nn = n0 + wn * 32 + t * 16 + cn;
#pragma unroll
    for (int s = 0; s < 4; ++s) {
      const int mb = m0 + wm * 64 + s * 16 + mh;
#pragma unroll
      for (int r = 0; r < 8; ++r) {
        const int m = mb + r;                 // slot
        if (idxp) {
          if (m < cnt) {
            float* po = out + (size_t)idxp[m] * Dn + nn;
            *po = *po + wtsp[m] * acc[s][t][r];
          }
        } else {
          out[(size_t)m * Dn + nn] = acc[s][t][r];
        }
      }
    }
  }
}

// ---------------------------------------------------------------------------
extern "C" void kernel_launch(void* const* d_in, const int* in_sizes, int n_in,
                              void* d_out, int out_size, void* d_ws, size_t ws_size,
                              hipStream_t stream) {
  const float* x    = (const float*)d_in[0];  // [4,2048,1024]
  const float* gw   = (const float*)d_in[1];  // [8,1024]
  const float* shgu = (const float*)d_in[2];  // [1876,1024]
  const float* shdn = (const float*)d_in[3];  // [1024,938]
  const float* exgu = (const float*)d_in[4];  // [8,1876,1024]
  const float* exdn = (const float*)d_in[5];  // [8,1024,938]
  float* out = (float*)d_out;                 // [4,2048,1024]

  // workspace: xb (16MB) | act (15MB) | idx (256KB) | wts (256KB) | cnt
  __bf16* xb  = (__bf16*)d_ws;
  __bf16* act = xb + (size_t)Tn * Dn;
  int*    idx = (int*)(act + (size_t)Tn * IPn);
  float*  wts = (float*)(idx + (size_t)En * Tn);
  int*    cnt = (int*)(wts + (size_t)En * Tn);

  init_counts_kernel<<<1, 32, 0, stream>>>(cnt);
  router_kernel<<<Tn / 8, 256, 0, stream>>>(x, gw, cnt, idx, wts);
  cast_kernel<<<(int)(((size_t)Tn * Dn) / 2048), 256, 0, stream>>>(x, xb);

  const dim3 gAct(IPn / 64,  Tn / 128);   // 15 x 64
  const dim3 gDn (Dn  / 128, Tn / 128);   //  8 x 64

  // shared expert: all tokens, identity mapping, writes out
  gemm_act_kernel <<<gAct, 256, 0, stream>>>(xb, shgu, act, nullptr, nullptr);
  gemm_down_kernel<<<gDn,  256, 0, stream>>>(act, shdn, out, nullptr, nullptr, nullptr);

  // routed experts on compacted token lists (blocks past count early-exit)
  for (int e = 0; e < En; ++e) {
    gemm_act_kernel <<<gAct, 256, 0, stream>>>(xb, exgu + (size_t)e * 2 * In * Dn,
                                               act, idx + (size_t)e * Tn, cnt + e);
    gemm_down_kernel<<<gDn,  256, 0, stream>>>(act, exdn + (size_t)e * Dn * In,
                                               out, idx + (size_t)e * Tn,
                                               wts + (size_t)e * Tn, cnt + e);
  }
}
